// ModulatedConv2d_54640573940174
// MI455X (gfx1250) — matrix operand order repaired and checked by reading, not compiled, not run on hardware
//
#include <hip/hip_runtime.h>

typedef __bf16 bf16;
typedef __attribute__((ext_vector_type(8)))  bf16  v8bf;
typedef __attribute__((ext_vector_type(16))) bf16  v16bf;
typedef __attribute__((ext_vector_type(8)))  float v8f;
typedef __attribute__((ext_vector_type(4)))  int   v4i;

#define AS1 __attribute__((address_space(1)))
#define AS3 __attribute__((address_space(3)))

#define B_    16
#define CIN_  256
#define COUT_ 256
#define HH    64
#define WW    64
#define SDIM_ 512
#define KD_   2304   // CIN*3*3, ordered (kh,kw)-major, cin-minor
#define HP    66
#define WP    66

#if defined(__gfx1250__) && __has_builtin(__builtin_amdgcn_global_load_async_to_lds_b128)
#define USE_ASYNC 1
#else
#define USE_ASYNC 0
#endif

// ---- 16B global -> LDS copy (async on CDNA5 if builtin available) ----------
__device__ __forceinline__ void cp16(const bf16* g, bf16* l) {
#if USE_ASYNC
  __builtin_amdgcn_global_load_async_to_lds_b128(
      (AS1 v4i*)(AS1 void*)g, (AS3 v4i*)(AS3 void*)l, 0, 0);
#else
  *(uint4*)l = *(const uint4*)g;
#endif
}
__device__ __forceinline__ void wait_cp() {
#if USE_ASYNC
#if __has_builtin(__builtin_amdgcn_s_wait_asynccnt)
  __builtin_amdgcn_s_wait_asynccnt(0);
#else
  asm volatile("s_wait_asynccnt 0x0" ::: "memory");
#endif
#endif
}

// ---- kernel 0: zero padded activation buffer -------------------------------
__global__ void zero_ws(uint4* p, size_t n16) {
  size_t i = (size_t)blockIdx.x * blockDim.x + threadIdx.x;
  size_t stride = (size_t)gridDim.x * blockDim.x;
  uint4 z{0u, 0u, 0u, 0u};
  for (; i < n16; i += stride) p[i] = z;
}

// ---- kernel 1: style projection s = style @ style_w + style_b --------------
__global__ __launch_bounds__(256)
void style_proj(const float* __restrict__ style, const float* __restrict__ sw,
                const float* __restrict__ sb, float* __restrict__ sp) {
  int b = blockIdx.x, c = threadIdx.x;
  float a = sb[c];
  for (int k = 0; k < SDIM_; ++k)
    a += style[(size_t)b * SDIM_ + k] * sw[(size_t)k * CIN_ + c];
  sp[b * CIN_ + c] = a;
}

// ---- kernel 2: modulate + demodulate + pack bf16 weights -------------------
// Wb[b][cout][k], k = (kh*3+kw)*256 + cin
__global__ __launch_bounds__(256)
void mod_weights(const float* __restrict__ wgt, const float* __restrict__ sp,
                 bf16* __restrict__ Wb) {
  __shared__ float wtmp[KD_];
  __shared__ float red[256];
  int cout = blockIdx.x, b = blockIdx.y;
  int cin = threadIdx.x;
  const float scale = 1.0f / 48.0f;  // 1/sqrt(CIN*3*3)
  float sv = sp[b * CIN_ + cin] * scale;
  float sq = 0.f;
  size_t base = ((size_t)cout * CIN_ + cin) * 9;
#pragma unroll
  for (int kk = 0; kk < 9; ++kk) {
    float wv = wgt[base + kk] * sv;
    wtmp[kk * CIN_ + cin] = wv;
    sq += wv * wv;
  }
  red[cin] = sq;
  __syncthreads();
  for (int off = 128; off > 0; off >>= 1) {
    if (cin < off) red[cin] += red[cin + off];
    __syncthreads();
  }
  float demod = rsqrtf(red[0] + 1e-8f);
  bf16* dst = Wb + ((size_t)b * COUT_ + cout) * KD_ + cin;
#pragma unroll
  for (int kk = 0; kk < 9; ++kk)
    dst[kk * CIN_] = (bf16)(wtmp[kk * CIN_ + cin] * demod);
}

// ---- kernel 3: NCHW f32 -> padded NHWC bf16 (LDS transpose) ----------------
__global__ __launch_bounds__(256)
void pack_x(const float* __restrict__ x, bf16* __restrict__ xh) {
  __shared__ float t[32 * 65];
  int tid = threadIdx.x;
  int cg = blockIdx.x, h = blockIdx.y, b = blockIdx.z;
  int cb = cg * 32;
  int col = tid & 63, r0 = tid >> 6;
#pragma unroll
  for (int i = 0; i < 8; ++i) {
    int row = r0 + i * 4;  // channel within group
    t[row * 65 + col] = x[(((size_t)b * CIN_ + cb + row) * HH + h) * WW + col];
  }
  __syncthreads();
#pragma unroll
  for (int j = 0; j < 8; ++j) {
    int idx = tid + j * 256;
    int ci = idx & 31, wp = idx >> 5;
    xh[(((size_t)b * HP + h + 1) * WP + (wp + 1)) * CIN_ + cb + ci] =
        (bf16)t[ci * 65 + wp];
  }
}

// ---- kernel 4: implicit-GEMM conv with WMMA bf16 ---------------------------
// D[b][cout 256][pix 4096] = Wb[b] (256 x 2304) x im2col(xh[b]) (2304 x 4096)
// Block tile: 128 cout x 128 pix (2 rows x 64 cols). 8 waves, each 64x32.
// LDS rows padded to 40 elems (80B) -> conflict-free fragment reads.
// A tile stored in natural k order; lane half h reads k = 16h..16h+15 (same for B).
__global__ __launch_bounds__(256)
void conv_wmma(const bf16* __restrict__ Wb, const bf16* __restrict__ xh,
               float* __restrict__ out) {
  __shared__ __align__(16) bf16 As[2][128 * 40];
  __shared__ __align__(16) bf16 Bs[2][128 * 40];
  const int tid  = threadIdx.x;
  const int lane = tid & 31;
  const int w    = tid >> 5;
  const int wm   = (w >> 2) * 64;  // wave cout offset in block tile
  const int wn   = (w & 3) * 32;   // wave pixel offset in block tile
  const int bb   = blockIdx.z;
  const int M0   = blockIdx.y * 128;
  const int N0   = blockIdx.x * 128;
  const int y0   = blockIdx.x * 2;

  const bf16* Abase = Wb + (size_t)(bb * COUT_ + M0) * KD_;

  v8f acc[4][2] = {};

  auto load_tiles = [&](int it, int buf) {
    const int k0 = it * 32;
    const int tap = k0 >> 8;          // (kh,kw) tap index 0..8
    const int cinb = k0 & 255;
    const int kh = tap / 3, kw = tap - kh * 3;
#pragma unroll
    for (int i = 0; i < 2; ++i) {     // A: 512 x 16B chunks, sequential
      int c = tid + i * 256;
      int m = c >> 2, p = c & 3;
      int s = ((p & 1) << 1) | (p >> 1);  // 0->0, 1->2, 2->1, 3->3
      cp16(Abase + (size_t)m * KD_ + k0 + p * 8, &As[buf][m * 40 + p * 8]);
    }
    const bf16* Bbase =
        xh + ((size_t)(bb * HP + y0 + kh) * WP + kw) * CIN_ + cinb;
#pragma unroll
    for (int i = 0; i < 2; ++i) {     // B: 512 x 16B chunks, sequential
      int c = tid + i * 256;
      int n = c >> 2, p = c & 3;
      int yy = n >> 6, xx = n & 63;
      cp16(Bbase + ((size_t)yy * WP + xx) * CIN_ + p * 8,
           &Bs[buf][n * 40 + p * 8]);
    }
  };

  load_tiles(0, 0);
  wait_cp();
  __syncthreads();

  const int r  = lane & 15;
  const int h2 = lane >> 4;
  int cur = 0;
  for (int it = 0; it < 72; ++it) {
    if (it + 1 < 72) load_tiles(it + 1, cur ^ 1);
    const bf16* Ab = &As[cur][0];
    const bf16* Bb = &Bs[cur][0];
    v16bf af[4], bfv[2];
#pragma unroll
    for (int ms = 0; ms < 4; ++ms) {
      int base = (wm + ms * 16 + r) * 40 + h2 * 16;
      v8bf lo = *(const v8bf*)(Ab + base);
      v8bf hi = *(const v8bf*)(Ab + base + 8);
      af[ms] = __builtin_shufflevector(lo, hi, 0, 1, 2, 3, 4, 5, 6, 7, 8, 9,
                                       10, 11, 12, 13, 14, 15);
    }
#pragma unroll
    for (int ns = 0; ns < 2; ++ns) {
      int base = (wn + ns * 16 + r) * 40 + h2 * 16;
      v8bf lo = *(const v8bf*)(Bb + base);
      v8bf hi = *(const v8bf*)(Bb + base + 8);
      bfv[ns] = __builtin_shufflevector(lo, hi, 0, 1, 2, 3, 4, 5, 6, 7, 8, 9,
                                        10, 11, 12, 13, 14, 15);
    }
#pragma unroll
    for (int ms = 0; ms < 4; ++ms)
#pragma unroll
      for (int ns = 0; ns < 2; ++ns)
        acc[ms][ns] = __builtin_amdgcn_wmma_f32_16x16x32_bf16(
            false, af[ms], false, bfv[ns], (short)0, acc[ms][ns], false, false);
    if (it + 1 < 72) {
      wait_cp();
      __syncthreads();
      cur ^= 1;
    }
  }

  // Epilogue: D 16x16 f32 layout -> lane n = lane&15, m = vgpr + (lane>>4)*8
  float* outb = out + (size_t)bb * COUT_ * (HH * WW);
#pragma unroll
  for (int ms = 0; ms < 4; ++ms) {
#pragma unroll
    for (int ns = 0; ns < 2; ++ns) {
      int cout0 = M0 + wm + ms * 16 + h2 * 8;
      int nn = N0 + wn + ns * 16 + r;
      int y = nn >> 6, xx = nn & 63;
      float* p = outb + ((size_t)cout0 * HH + y) * WW + xx;
#pragma unroll
      for (int q = 0; q < 8; ++q) p[(size_t)q * (HH * WW)] = acc[ms][ns][q];
    }
  }
}

// ---- launcher --------------------------------------------------------------
extern "C" void kernel_launch(void* const* d_in, const int* in_sizes, int n_in,
                              void* d_out, int out_size, void* d_ws,
                              size_t ws_size, hipStream_t stream) {
  const float* x  = (const float*)d_in[0];
  const float* st = (const float*)d_in[1];
  const float* wg = (const float*)d_in[2];
  const float* sw = (const float*)d_in[3];
  const float* sb = (const float*)d_in[4];
  float* out = (float*)d_out;

  char* ws = (char*)d_ws;
  float* sp = (float*)ws;                                   // 16 KB
  bf16* Wb = (bf16*)(ws + 16384);                           // 18.87 MB
  bf16* xh = (bf16*)(ws + 16384 + (size_t)B_ * COUT_ * KD_ * 2);  // 35.7 MB

  size_t xh16 = (size_t)B_ * HP * WP * CIN_ * 2 / 16;
  zero_ws<<<2048, 256, 0, stream>>>((uint4*)xh, xh16);
  style_proj<<<B_, 256, 0, stream>>>(st, sw, sb, sp);
  mod_weights<<<dim3(COUT_, B_), 256, 0, stream>>>(wg, sp, Wb);
  pack_x<<<dim3(8, HH, B_), 256, 0, stream>>>(x, xh);
  conv_wmma<<<dim3(32, 2, B_), 256, 0, stream>>>(Wb, xh, out);
}
